// CosineAttention_670014898398
// MI455X (gfx1250) — matrix operand order repaired and checked by reading, not compiled, not run on hardware
//
#include <hip/hip_runtime.h>
#include <hip/hip_bf16.h>

// CDNA5 / gfx1250 wave32 WMMA implementation of cosine-attention column means.
// Stage 1: P = rownorm(x @ W) in bf16        (v_wmma_f32_16x16x32_bf16)
// Stage 2: out[b,n] = mean_s sigmoid(P P^T)  (v_wmma_f32_16x16x32_bf16,
//          key block staged via global_load_async_to_lds_b128 + ASYNCcnt)

typedef __attribute__((ext_vector_type(16))) __bf16 v16bf;
typedef __attribute__((ext_vector_type(8)))  float  v8f;

union BF16Frag {
  v16bf v;
  unsigned short u[16];
  uint4 q[2];
};

__device__ __forceinline__ unsigned short f2bf(float f) {
  // round-to-nearest-even f32 -> bf16
  unsigned int x = __float_as_uint(f);
  unsigned int r = x + 0x7FFFu + ((x >> 16) & 1u);
  return (unsigned short)(r >> 16);
}

__device__ __forceinline__ v8f wmma_bf16(const BF16Frag& a, const BF16Frag& b, v8f c) {
  return __builtin_amdgcn_wmma_f32_16x16x32_bf16(
      /*neg_a=*/false, a.v, /*neg_b=*/false, b.v,
      /*c_mod=*/(short)0, c, /*reuse_a=*/false, /*reuse_b=*/false);
}

// Async global->LDS copy (16 B per lane), tracked by ASYNCcnt.
__device__ __forceinline__ void async_load_b128(unsigned lds_off, const void* gptr) {
  asm volatile("global_load_async_to_lds_b128 %0, %1, off"
               :: "v"(lds_off), "v"((unsigned long long)(uintptr_t)gptr)
               : "memory");
}
__device__ __forceinline__ void wait_async0() {
  asm volatile("s_wait_asynccnt 0x0" ::: "memory");
}

// ---------------------------------------------------------------------------
// Kernel 0: Wt[e][k] = bf16(W[k][e])  (1024x1024, LDS-tiled transpose)
// ---------------------------------------------------------------------------
__global__ void k_wt(const float* __restrict__ W, unsigned short* __restrict__ Wt) {
  __shared__ float tile[32][33];
  const int tx = threadIdx.x & 31;
  const int ty = threadIdx.x >> 5;          // 0..7
  const int k0 = blockIdx.x * 32;
  const int e0 = blockIdx.y * 32;
#pragma unroll
  for (int i = 0; i < 32; i += 8)
    tile[ty + i][tx] = W[(size_t)(k0 + ty + i) * 1024 + e0 + tx];
  __syncthreads();
#pragma unroll
  for (int i = 0; i < 32; i += 8)
    Wt[(size_t)(e0 + ty + i) * 1024 + k0 + tx] = f2bf(tile[tx][ty + i]);
}

// ---------------------------------------------------------------------------
// Kernel 1: P = rownorm(x @ W), bf16 output.
// One workgroup (8 waves) per 16-row block of the [32768 x 1024] problem;
// wave w owns columns [w*128, w*128+128).
// ---------------------------------------------------------------------------
__global__ void __launch_bounds__(256)
k_proj(const float* __restrict__ x, const unsigned short* __restrict__ Wt,
       unsigned short* __restrict__ P) {
  const int lane = threadIdx.x & 31;
  const int wave = threadIdx.x >> 5;
  const int l16  = lane & 15;
  const int half = lane >> 4;               // lane half (K-offset select)
  const int row0 = blockIdx.x * 16;

  const float* xr = x + (size_t)(row0 + l16) * 1024;

  auto loadA = [&](BF16Frag& a, int kk) {
    // 16-bit A 16x32 layout: two 8-elem K segments {kbA, kbA+16}.
    const int kbA = kk + half * 8;
    const float4 a0 = *(const float4*)(xr + kbA);
    const float4 a1 = *(const float4*)(xr + kbA + 4);
    const float4 a2 = *(const float4*)(xr + kbA + 16);
    const float4 a3 = *(const float4*)(xr + kbA + 20);
    const float fa[16] = {a0.x, a0.y, a0.z, a0.w, a1.x, a1.y, a1.z, a1.w,
                          a2.x, a2.y, a2.z, a2.w, a3.x, a3.y, a3.z, a3.w};
#pragma unroll
    for (int i = 0; i < 16; ++i) a.u[i] = f2bf(fa[i]);
  };

  v8f acc[8];
  const v8f vzero = {0.f, 0.f, 0.f, 0.f, 0.f, 0.f, 0.f, 0.f};
#pragma unroll
  for (int t = 0; t < 8; ++t) acc[t] = vzero;

  BF16Frag aCur;
  loadA(aCur, 0);

  for (int kk = 0; kk < 1024; kk += 32) {
    // Batch all 8 B fragments (distinct regs -> clause of b128 loads).
    const int kbB = kk + half * 8;          // B: two 8-elem K segments {kbB, kbB+16}
    BF16Frag bf[8];
#pragma unroll
    for (int t = 0; t < 8; ++t) {
      const unsigned short* wr =
          Wt + (size_t)((wave * 8 + t) * 16 + l16) * 1024 + kbB;
      bf[t].q[0] = *(const uint4*)(wr);
      bf[t].q[1] = *(const uint4*)(wr + 16);
    }
    // Prefetch + convert next A while WMMAs run.
    BF16Frag aNxt;
    if (kk + 32 < 1024) loadA(aNxt, kk + 32);
#pragma unroll
    for (int t = 0; t < 8; ++t) acc[t] = wmma_bf16(aCur, bf[t], acc[t]);
    if (kk + 32 < 1024) aCur = aNxt;
  }

  // Row sum-of-squares; C layout: VGPR m <-> row (half*8 + m), col = l16.
  __shared__ float ssq[16];
  if (threadIdx.x < 16) ssq[threadIdx.x] = 0.f;
  __syncthreads();

  float ss[8];
#pragma unroll
  for (int m = 0; m < 8; ++m) {
    float s = 0.f;
#pragma unroll
    for (int t = 0; t < 8; ++t) s += acc[t][m] * acc[t][m];
    ss[m] = s;
  }
#pragma unroll
  for (int off = 1; off < 16; off <<= 1) {
#pragma unroll
    for (int m = 0; m < 8; ++m) ss[m] += __shfl_xor(ss[m], off, 32);
  }
  if (l16 == 0) {
#pragma unroll
    for (int m = 0; m < 8; ++m) atomicAdd(&ssq[half * 8 + m], ss[m]);
  }
  __syncthreads();

  float inv[8];
#pragma unroll
  for (int m = 0; m < 8; ++m) {
    const float nrm = __builtin_sqrtf(ssq[half * 8 + m]);
    inv[m] = 1.0f / fmaxf(nrm, 1e-12f);     // torch F.normalize eps
  }

#pragma unroll
  for (int t = 0; t < 8; ++t) {
    const int n = (wave * 8 + t) * 16 + l16;
#pragma unroll
    for (int m = 0; m < 8; ++m) {
      const int r = row0 + half * 8 + m;
      P[(size_t)r * 1024 + n] = f2bf(acc[t][m] * inv[m]);
    }
  }
}

// ---------------------------------------------------------------------------
// Kernel 2: out[b,n] += (1/4096) * sum_q sigmoid(P_b[q,:] . P_b[n,:])
// grid = (32 key blocks, 32 query blocks, 8 batches); wave w: 16 queries.
// Key block K-slices are staged to LDS with async copies, double buffered.
// ---------------------------------------------------------------------------
#define SK_STRIDE 40   // ushorts per LDS row (80 B): conflict-free b128 reads

__global__ void __launch_bounds__(256)
k_sim(const unsigned short* __restrict__ P, float* __restrict__ out) {
  __shared__ unsigned short sK[2][128][SK_STRIDE];   // 2 x 128 rows x 32 K (padded)
  __shared__ float colsum[128];

  const int tid  = threadIdx.x;
  const int lane = tid & 31;
  const int wave = tid >> 5;
  const int l16  = lane & 15;
  const int half = lane >> 4;
  const int b    = blockIdx.z;
  const unsigned short* Pb = P + (size_t)b * 4096 * 1024;

  const int n0 = blockIdx.x * 128;                   // key block (shared by waves)
  const int q0 = blockIdx.y * 128 + wave * 16;       // this wave's queries

  for (int i = tid; i < 128; i += 256) colsum[i] = 0.f;

  const unsigned sK_base = (unsigned)(uintptr_t)(&sK[0][0][0]);
  const int cr = tid >> 1;                           // copy row 0..127
  const int cs = tid & 1;                            // 32-byte segment 0..1

  // Issue async copy of the [128 x 32] bf16 key slice at K=kk into buffer buf.
  auto stage = [&](int buf, int kk) {
    const unsigned short* g = Pb + (size_t)(n0 + cr) * 1024 + kk + cs * 16;
    const unsigned loff =
        sK_base + (unsigned)(buf * (128 * SK_STRIDE * 2) + cr * (SK_STRIDE * 2) + cs * 32);
    async_load_b128(loff,      g);
    async_load_b128(loff + 16, g + 8);
  };

  const unsigned short* qr = Pb + (size_t)(q0 + l16) * 1024;
  auto loadA = [&](BF16Frag& a, int kk) {
    const int kbA = kk + half * 8;
    a.q[0] = *(const uint4*)(qr + kbA);
    a.q[1] = *(const uint4*)(qr + kbA + 16);
  };

  v8f acc[8];
  const v8f vzero = {0.f, 0.f, 0.f, 0.f, 0.f, 0.f, 0.f, 0.f};
#pragma unroll
  for (int t = 0; t < 8; ++t) acc[t] = vzero;

  // Prologue: stage buffer 0, load first A fragment.
  stage(0, 0);
  BF16Frag aCur;
  loadA(aCur, 0);
  wait_async0();
  __syncthreads();

  for (int s = 0; s < 32; ++s) {
    const int kk = s * 32;
    const int buf = s & 1;

    // Kick off next slice + next A while this step computes.
    if (s + 1 < 32) stage(buf ^ 1, kk + 32);
    BF16Frag aNxt;
    if (s + 1 < 32) loadA(aNxt, kk + 32);

    // Read all 8 B fragments from LDS (clause of ds_load_b128).
    const unsigned short* kbase = &sK[buf][0][0];
    BF16Frag bf[8];
#pragma unroll
    for (int t = 0; t < 8; ++t) {
      const unsigned short* src = kbase + (t * 16 + l16) * SK_STRIDE + half * 8;
      bf[t].q[0] = *(const uint4*)(src);
      bf[t].q[1] = *(const uint4*)(src + 16);
    }
#pragma unroll
    for (int t = 0; t < 8; ++t) acc[t] = wmma_bf16(aCur, bf[t], acc[t]);

    if (s + 1 < 32) {
      aCur = aNxt;
      wait_async0();        // my async writes for buf^1 are done
      __syncthreads();      // everyone's writes done; everyone done reading buf
    }
  }

  // sigmoid + per-column (key) partial sums over this wave's 16 query rows.
  __syncthreads();          // colsum[] init visible / LDS reuse safe
#pragma unroll
  for (int t = 0; t < 8; ++t) {
    float cp = 0.f;
#pragma unroll
    for (int m = 0; m < 8; ++m) {
      const float s = acc[t][m];
      cp += 1.0f / (1.0f + __expf(-s));
    }
    cp += __shfl_xor(cp, 16, 32);            // combine row halves
    if (half == 0) atomicAdd(&colsum[t * 16 + l16], cp);
  }
  __syncthreads();

  if (tid < 128) {
    atomicAdd(out + (size_t)b * 4096 + n0 + tid, colsum[tid] * (1.0f / 4096.0f));
  }
}

__global__ void k_zero(float* __restrict__ out, int n) {
  const int i = blockIdx.x * blockDim.x + threadIdx.x;
  if (i < n) out[i] = 0.f;
}

// ---------------------------------------------------------------------------
extern "C" void kernel_launch(void* const* d_in, const int* in_sizes, int n_in,
                              void* d_out, int out_size, void* d_ws, size_t ws_size,
                              hipStream_t stream) {
  const float* x = (const float*)d_in[0];   // [8,4096,1024] f32
  const float* W = (const float*)d_in[1];   // [1024,1024]   f32
  float* out = (float*)d_out;               // [8,1,4096]    f32

  // Workspace layout: Wt bf16 (2 MB) @ 0, P bf16 (64 MB) @ 2 MB.
  unsigned short* Wt = (unsigned short*)d_ws;
  unsigned short* P  = (unsigned short*)((char*)d_ws + (size_t)(2u << 20));

  k_wt  <<<dim3(32, 32),    256, 0, stream>>>(W, Wt);
  k_proj<<<dim3(2048),      256, 0, stream>>>(x, Wt, P);
  k_zero<<<dim3(128),       256, 0, stream>>>(out, 8 * 4096);
  k_sim <<<dim3(32, 32, 8), 256, 0, stream>>>(P, out);
}
